// HAT_5291399708853
// MI455X (gfx1250) — matrix operand order repaired and checked by reading, not compiled, not run on hardware
//
#include <hip/hip_runtime.h>

// ---------------------------------------------------------------------------
// HAT windowed attention, fully fused for MI455X (gfx1250, wave32, WMMA).
// One workgroup per window b. All GEMMs via v_wmma_f32_16x16x32_f16.
// ---------------------------------------------------------------------------

#define NTOK 53        // 7*7 + 2*2 tokens
#define NPAD 64        // padded token count (4 x 16 tiles)
#define CDIM 256
#define HEADS 8
#define HD 32

typedef __attribute__((ext_vector_type(16))) _Float16 v16h;
typedef __attribute__((ext_vector_type(8)))  _Float16 v8h;
typedef __attribute__((ext_vector_type(4)))  _Float16 v4h;
typedef __attribute__((ext_vector_type(8)))  float    v8f;
typedef __attribute__((ext_vector_type(4)))  float    v4f;

union F16Frag { v16h v; v8h h[2]; _Float16 e[16]; };

__device__ __forceinline__ v8f wmma32(v16h a, v16h b, v8f c) {
  // D = A(16x32 f16) * B(32x16 f16) + C(16x16 f32)
  return __builtin_amdgcn_wmma_f32_16x16x32_f16(false, a, false, b,
                                                (short)0, c, false, false);
}

// A-matrix fragment (16x32 f16) from row-major [row][k] storage.
// ISA layout: lanes 0-15 rows M=lane, K 0-7 in V0-3 / 16-23 in V4-7;
//             lanes 16-31 same rows, K 8-15 / 24-31.
__device__ __forceinline__ v16h loadA(const _Float16* base, int stride,
                                      int row0, int k0, int lane) {
  int m  = row0 + (lane & 15);
  int kb = (lane < 16) ? 0 : 8;
  const _Float16* p = base + m * stride + k0;
  F16Frag f;
  f.h[0] = *(const v8h*)(p + kb);        // K = k0+kb .. +7
  f.h[1] = *(const v8h*)(p + 16 + kb);   // K = k0+16+kb .. +7
  return f.v;
}

// B-matrix fragment (32x16 f16), element B[k][n] taken from base[n*stride+k]
// (i.e. N is the strided dim, K contiguous: weight rows, K^T rows, V^T rows).
// ISA layout: lanes 0-15 K 0-7 / 16-23, lanes 16-31 K 8-15 / 24-31 (matches loadA).
__device__ __forceinline__ v16h loadB_rowmajor(const _Float16* base, int stride,
                                               int n0, int k0, int lane) {
  int n  = n0 + (lane & 15);
  int kb = (lane < 16) ? 0 : 8;
  const _Float16* p = base + n * stride + k0 + kb;
  F16Frag f;
  f.h[0] = *(const v8h*)p;
  f.h[1] = *(const v8h*)(p + 16);
  return f.v;
}

// ---------------------------------------------------------------------------
// Prep: f32->f16 weight conversion + padded bias/mask matrix (8 x 64 x 64).
// ---------------------------------------------------------------------------
__global__ void hat_prep(const float* __restrict__ qkv_w,
                         const float* __restrict__ proj_w,
                         const float* __restrict__ tbl,
                         _Float16* __restrict__ qw16,
                         _Float16* __restrict__ pw16,
                         float* __restrict__ biasPad) {
  int i = blockIdx.x * blockDim.x + threadIdx.x;
  if (i < 768 * 256) qw16[i] = (_Float16)qkv_w[i];
  if (i < 256 * 256) pw16[i] = (_Float16)proj_w[i];
  if (i < HEADS * NPAD * NPAD) {
    int h = i >> 12, rem = i & 4095, n = rem >> 6, m = rem & 63;
    float v = 0.f;
    if (m >= NTOK) {
      v = -1e30f;                       // mask padded key columns
    } else if (n >= 4 && n < NTOK && m >= 4) {
      int ii = n - 4, jj = m - 4;       // window-local token indices
      int ri = ii / 7, ci = ii % 7, rj = jj / 7, cj = jj % 7;
      int idx = (ri - rj + 6) * 13 + (ci - cj + 6);
      v = tbl[idx * HEADS + h];
    }
    biasPad[i] = v;
  }
}

// ---------------------------------------------------------------------------
// Fused attention kernel: one block per window, 8 waves.
// ---------------------------------------------------------------------------
__global__ __launch_bounds__(256, 1) void hat_fused(
    const float* __restrict__ x, const float* __restrict__ qkv_b,
    const float* __restrict__ proj_b, const _Float16* __restrict__ qw16,
    const _Float16* __restrict__ pw16, const float* __restrict__ biasPad,
    float* __restrict__ out) {

  constexpr int XS_STRIDE  = CDIM + 8;   // 264 f16 (bank skew)
  constexpr int QKV_STRIDE = 512 + 8;    // 520 f16 (Q+K only, token-major)
  constexpr int VT_STRIDE  = 64 + 8;     // 72 f16  (V^T: [d][m], m contiguous)
  constexpr int PST_STRIDE = 64 + 8;     // 72 f16
  constexpr int OUT_STRIDE = CDIM + 8;   // 264 f16

  // Overlapped LDS layout (152 KB; two blocks fit in a 320 KB WGP):
  //   [0, 66560)         qkvs : 64 x 520 f16           (phase 1-2)  Q|K
  //   [66560, 103424)    vts  : 256 x 72 f16           (phase 1-2)  V^T
  //   [103424, 137216)   xs   : 64 x 264 f16           (phase 0-1)
  //   [103424, 121856)   pstg : 8 waves x 16 x 72 f16  (phase 2)
  //   [121856, 155648)   outs : 64 x 264 f16           (phase 2-3)
  __shared__ alignas(16) char smem[155648];
  _Float16* qkvs = (_Float16*)smem;
  _Float16* vts  = (_Float16*)(smem + 66560);
  _Float16* xs   = (_Float16*)(smem + 103424);
  _Float16* pstg = (_Float16*)(smem + 103424);
  _Float16* outs = (_Float16*)(smem + 121856);

  const int tid  = threadIdx.x;
  const int w    = tid >> 5;
  const int lane = tid & 31;
  const int b    = blockIdx.x;
  const int colb = lane & 15;
  const int rb   = (lane < 16) ? 0 : 8;   // C/D row offset per half-wave

  const v8f vzero = {0.f, 0.f, 0.f, 0.f, 0.f, 0.f, 0.f, 0.f};

  // ---- phase 0: stage x -> LDS f16 (float4 loads), zero padded rows -----
  const float* xb = x + (size_t)b * (NTOK * CDIM);
  const v4f* x4 = (const v4f*)xb;
  for (int i = tid; i < NTOK * (CDIM / 4); i += 256) {
    v4f d = x4[i];
    int n = i >> 6, c4 = (i & 63) * 4;
    v4h hv = {(_Float16)d[0], (_Float16)d[1], (_Float16)d[2], (_Float16)d[3]};
    *(v4h*)(xs + n * XS_STRIDE + c4) = hv;
  }
  {
    const v4h hz = {(_Float16)0.f, (_Float16)0.f, (_Float16)0.f, (_Float16)0.f};
    for (int i = tid; i < (NPAD - NTOK) * (CDIM / 4); i += 256) {
      int n = NTOK + (i >> 6), c4 = (i & 63) * 4;
      *(v4h*)(xs + n * XS_STRIDE + c4) = hz;
    }
  }
  __syncthreads();

  // ---- phase 1: qkv = x @ qkv_w^T + qkv_b  (64 x 768) -------------------
  // Q,K tiles (cols 0..511) -> qkvs token-major; V tiles (512..767) -> vts
  // transposed [d][m] so P@V B-fragments are contiguous b128 loads.
  for (int t = 0; t < 6; ++t) {
    int col0 = (w + t * 8) * 16;           // output column tile
    int ntj  = w + (t + 1) * 8;
    if (ntj < 48) __builtin_prefetch(qw16 + (ntj * 16) * CDIM, 0, 1);
    v8f acc[4] = {vzero, vzero, vzero, vzero};
    #pragma unroll
    for (int ks = 0; ks < 8; ++ks) {
      v16h bf = loadB_rowmajor(qw16, CDIM, col0, ks * 32, lane);
      #pragma unroll
      for (int mi = 0; mi < 4; ++mi) {
        v16h af = loadA(xs, XS_STRIDE, mi * 16, ks * 32, lane);
        acc[mi] = wmma32(af, bf, acc[mi]);
      }
    }
    float bj = qkv_b[col0 + colb];
    if (col0 < 512) {                       // uniform branch (t < 4)
      #pragma unroll
      for (int mi = 0; mi < 4; ++mi)
        #pragma unroll
        for (int r = 0; r < 8; ++r)
          qkvs[(mi * 16 + rb + r) * QKV_STRIDE + col0 + colb] =
              (_Float16)(acc[mi][r] + bj);
    } else {                                // V: transposed, vectorized store
      int d = col0 + colb - 512;            // 0..255
      #pragma unroll
      for (int mi = 0; mi < 4; ++mi) {
        v8h hv;
        #pragma unroll
        for (int r = 0; r < 8; ++r) hv[r] = (_Float16)(acc[mi][r] + bj);
        *(v8h*)(vts + d * VT_STRIDE + mi * 16 + rb) = hv;
      }
    }
  }
  __syncthreads();

  // ---- phase 2: attention, wave w owns head w ---------------------------
  {
    const int h = w;
    const float scale = 0.17677669529663687f;   // 1/sqrt(32)
    const float* bp = biasPad + h * (NPAD * NPAD);
    const _Float16* k_base  = qkvs + 256 + h * HD;        // K, token rows
    const _Float16* vt_base = vts + (h * HD) * VT_STRIDE; // V^T, d rows
    _Float16* ps = pstg + w * (16 * PST_STRIDE);

    // V^T as B-fragments: n=d (strided rows), K=m contiguous -> b128 loads.
    v16h Bv[2][2];
    #pragma unroll
    for (int k2 = 0; k2 < 2; ++k2)
      #pragma unroll
      for (int nj = 0; nj < 2; ++nj)
        Bv[k2][nj] = loadB_rowmajor(vt_base, VT_STRIDE, nj * 16, k2 * 32, lane);

    for (int mi = 0; mi < 4; ++mi) {
      v16h Aq = loadA(qkvs, QKV_STRIDE, mi * 16, h * HD, lane);
      v8f S[4];
      #pragma unroll
      for (int ni = 0; ni < 4; ++ni) {
        v16h Bk = loadB_rowmajor(k_base, QKV_STRIDE, ni * 16, 0, lane);
        S[ni] = wmma32(Aq, Bk, vzero);
      }
      // scale + relative-position bias (mask folded in as -1e30)
      #pragma unroll
      for (int ni = 0; ni < 4; ++ni)
        #pragma unroll
        for (int r = 0; r < 8; ++r) {
          int row = mi * 16 + rb + r;
          int col = ni * 16 + colb;
          S[ni][r] = S[ni][r] * scale + bp[row * NPAD + col];
        }
      // row softmax: 16-lane butterfly (C layout: row's cols across a half-wave)
      #pragma unroll
      for (int r = 0; r < 8; ++r) {
        float mx = fmaxf(fmaxf(S[0][r], S[1][r]), fmaxf(S[2][r], S[3][r]));
        #pragma unroll
        for (int off = 1; off < 16; off <<= 1)
          mx = fmaxf(mx, __shfl_xor(mx, off, 32));
        float e0 = __expf(S[0][r] - mx);
        float e1 = __expf(S[1][r] - mx);
        float e2 = __expf(S[2][r] - mx);
        float e3 = __expf(S[3][r] - mx);
        float sum = (e0 + e1) + (e2 + e3);
        #pragma unroll
        for (int off = 1; off < 16; off <<= 1)
          sum += __shfl_xor(sum, off, 32);
        float inv = 1.f / sum;
        S[0][r] = e0 * inv; S[1][r] = e1 * inv;
        S[2][r] = e2 * inv; S[3][r] = e3 * inv;
      }
      // stage P (C layout -> LDS) and reload as A fragments (same-wave LDS
      // ops are in order; private per-wave staging buffer, no barrier needed)
      #pragma unroll
      for (int ni = 0; ni < 4; ++ni)
        #pragma unroll
        for (int r = 0; r < 8; ++r)
          ps[(rb + r) * PST_STRIDE + ni * 16 + colb] = (_Float16)S[ni][r];
      v16h Ap0 = loadA(ps, PST_STRIDE, 0, 0, lane);
      v16h Ap1 = loadA(ps, PST_STRIDE, 0, 32, lane);
      #pragma unroll
      for (int nj = 0; nj < 2; ++nj) {
        v8f o = wmma32(Ap0, Bv[0][nj], vzero);
        o = wmma32(Ap1, Bv[1][nj], o);
        #pragma unroll
        for (int r = 0; r < 8; ++r)
          outs[(mi * 16 + rb + r) * OUT_STRIDE + h * HD + nj * 16 + colb] =
              (_Float16)o[r];
      }
    }
  }
  __syncthreads();

  // ---- phase 3: out = attn_out @ proj_w^T + proj_b ----------------------
  for (int t = 0; t < 2; ++t) {
    int col0 = (w * 2 + t) * 16;
    if (t == 0) __builtin_prefetch(pw16 + (w * 2 + 1) * 16 * CDIM, 0, 1);
    v8f acc[4] = {vzero, vzero, vzero, vzero};
    #pragma unroll
    for (int ks = 0; ks < 8; ++ks) {
      v16h bf = loadB_rowmajor(pw16, CDIM, col0, ks * 32, lane);
      #pragma unroll
      for (int mi = 0; mi < 4; ++mi) {
        v16h af = loadA(outs, OUT_STRIDE, mi * 16, ks * 32, lane);
        acc[mi] = wmma32(af, bf, acc[mi]);
      }
    }
    float bj = proj_b[col0 + colb];
    #pragma unroll
    for (int mi = 0; mi < 4; ++mi)
      #pragma unroll
      for (int r = 0; r < 8; ++r) {
        int n = mi * 16 + rb + r;
        if (n < NTOK)
          out[((size_t)b * NTOK + n) * CDIM + col0 + colb] = acc[mi][r] + bj;
      }
  }
}

// ---------------------------------------------------------------------------
extern "C" void kernel_launch(void* const* d_in, const int* in_sizes, int n_in,
                              void* d_out, int out_size, void* d_ws,
                              size_t ws_size, hipStream_t stream) {
  const float* x      = (const float*)d_in[0];   // (B, 53, 256)
  const float* qkv_w  = (const float*)d_in[1];   // (768, 256)
  const float* qkv_b  = (const float*)d_in[2];   // (768,)
  const float* proj_w = (const float*)d_in[3];   // (256, 256)
  const float* proj_b = (const float*)d_in[4];   // (256,)
  const float* tbl    = (const float*)d_in[5];   // (169, 8)

  char* ws = (char*)d_ws;
  _Float16* qw16   = (_Float16*)ws;              // 393216 B
  _Float16* pw16   = (_Float16*)(ws + 393216);   // 131072 B
  float*    biasPd = (float*)(ws + 524288);      // 131072 B

  int B = in_sizes[0] / (NTOK * CDIM);           // 4096

  hat_prep<<<768, 256, 0, stream>>>(qkv_w, proj_w, tbl, qw16, pw16, biasPd);
  hat_fused<<<B, 256, 0, stream>>>(x, qkv_b, proj_b, qw16, pw16, biasPd,
                                   (float*)d_out);
  (void)n_in; (void)out_size; (void)ws_size;
}
